// GraphMHC_28999619183081
// MI455X (gfx1250) — matrix operand order repaired and checked by reading, not compiled, not run on hardware
//
#include <hip/hip_runtime.h>
#include <hip/hip_bf16.h>
#include <math.h>

#define N_NODES  50000
#define N_EDGES  200000
#define N_GRAPHS 64
#define IN_CH    31
#define EDGE_DIM 12
#define CH       256
#define HEADS    8
#define HC       (HEADS * CH)          /* 2048 */
#define KSIZE    9
#define CONV_OUT 8
#define CONV_LEN (2 * CH - KSIZE + 1)  /* 504 */
#define POOL_LEN (CONV_LEN / 2)        /* 252 */
#define EPSV     1e-5f
#define MP       50048                 /* N_NODES padded to multiple of 64 */

typedef __attribute__((ext_vector_type(16))) _Float16 v16h;
typedef __attribute__((ext_vector_type(8)))  float    v8f;

union Frag16 { v16h v; _Float16 h[16]; unsigned u[8]; };

// ---------- order-preserving float<->uint for atomicMax ----------
__device__ __forceinline__ unsigned enc_f32(float f) {
    unsigned u = __float_as_uint(f);
    return (u & 0x80000000u) ? ~u : (u | 0x80000000u);
}
__device__ __forceinline__ float dec_f32(unsigned k) {
    unsigned u = (k & 0x80000000u) ? (k & 0x7FFFFFFFu) : ~k;
    return __uint_as_float(u);
}

// ---------- operand pack kernels (fp32 -> f16 in WMMA fragment order) ----------
// A[M,K] -> Apack fragments: frag index = (mt*KT + kt), 32 lanes x 16 halfs contiguous.
// Per ISA 7.12.2 (16-bit A 16x32): lanes 0-15 hold K {0..7,16..23}, lanes 16-31 {8..15,24..31},
// two consecutive K per VGPR.
__global__ void k_pack_a(const float* __restrict__ src, _Float16* __restrict__ dst,
                         int M, int Mp, int K, int KT) {
    int id = blockIdx.x * blockDim.x + threadIdx.x;     // (mt*KT+kt)*32 + lane
    int total = (Mp >> 4) * KT * 32;
    if (id >= total) return;
    int lane = id & 31, frag = id >> 5;
    int kt = frag % KT, mt = frag / KT;
    int halfSel = lane >> 4, lm = lane & 15;
    int row = mt * 16 + lm;
    Frag16 f;
#pragma unroll
    for (int v = 0; v < 8; ++v) {
        int kk = (v < 4) ? (halfSel * 8 + 2 * v) : (16 + halfSel * 8 + 2 * (v - 4));
        int k = kt * 32 + kk;
        float x0 = (row < M && k     < K) ? src[(long)row * K + k]     : 0.0f;
        float x1 = (row < M && k + 1 < K) ? src[(long)row * K + k + 1] : 0.0f;
        f.h[2 * v]     = (_Float16)x0;
        f.h[2 * v + 1] = (_Float16)x1;
    }
    *(v16h*)(dst + (size_t)id * 16) = f.v;
}
// B[K,N] -> Bpack fragments: frag index = (kt*NT + nt).
// Same K order as A: lanes 0-15 hold K {0..7,16..23}, lanes 16-31 {8..15,24..31}, 2 K per VGPR; N = lane%16.
__global__ void k_pack_b(const float* __restrict__ src, _Float16* __restrict__ dst,
                         int K, int KT, int N) {
    int id = blockIdx.x * blockDim.x + threadIdx.x;     // (kt*NT+nt)*32 + lane
    int NT = N >> 4;
    int total = KT * NT * 32;
    if (id >= total) return;
    int lane = id & 31, frag = id >> 5;
    int nt = frag % NT, kt = frag / NT;
    int halfSel = lane >> 4, lm = lane & 15;
    int n = nt * 16 + lm;
    Frag16 f;
#pragma unroll
    for (int v = 0; v < 8; ++v) {
        int kk = (v < 4) ? (halfSel * 8 + 2 * v) : (16 + halfSel * 8 + 2 * (v - 4));
        int k = kt * 32 + kk;
        f.h[2 * v]     = (k     < K) ? (_Float16)src[(long)k * N + n]       : (_Float16)0.0f;
        f.h[2 * v + 1] = (k + 1 < K) ? (_Float16)src[(long)(k + 1) * N + n] : (_Float16)0.0f;
    }
    *(v16h*)(dst + (size_t)id * 16) = f.v;
}

// ---------- WMMA GEMM on pre-packed fragments ----------
// C[M,N](f16) = A x B, fp32 acc. grid = (N/64, Mp/64), block = 128 (4 waves);
// wave w owns m-tile (by*4+w); each iteration: 1 A frag + 4 B frags (32B contiguous per lane).
__global__ void k_wmma_gemm(const _Float16* __restrict__ Apack, const _Float16* __restrict__ Bpack,
                            _Float16* __restrict__ C, int M, int KT, int N) {
    const int lane = threadIdx.x & 31;
    const int wave = threadIdx.x >> 5;
    const int NT = N >> 4;
    const int mt  = blockIdx.y * 4 + wave;
    const int nt0 = blockIdx.x * 4;
    const int halfSel = lane >> 4, lm = lane & 15;

    v8f acc0 = {}, acc1 = {}, acc2 = {}, acc3 = {};
    const _Float16* Ab = Apack + ((size_t)mt * KT) * 512 + lane * 16;
    const _Float16* Bb = Bpack + ((size_t)nt0) * 512 + lane * 16;

    for (int kt = 0; kt < KT; ++kt) {
        v16h a  = *(const v16h*)(Ab + (size_t)kt * 512);
        const _Float16* Bk = Bb + (size_t)kt * NT * 512;
        v16h b0 = *(const v16h*)(Bk + 0 * 512);
        v16h b1 = *(const v16h*)(Bk + 1 * 512);
        v16h b2 = *(const v16h*)(Bk + 2 * 512);
        v16h b3 = *(const v16h*)(Bk + 3 * 512);
        acc0 = __builtin_amdgcn_wmma_f32_16x16x32_f16(false, a, false, b0, (short)0, acc0, false, false);
        acc1 = __builtin_amdgcn_wmma_f32_16x16x32_f16(false, a, false, b1, (short)0, acc1, false, false);
        acc2 = __builtin_amdgcn_wmma_f32_16x16x32_f16(false, a, false, b2, (short)0, acc2, false, false);
        acc3 = __builtin_amdgcn_wmma_f32_16x16x32_f16(false, a, false, b3, (short)0, acc3, false, false);
    }
    // D layout: N = lane%16, M = v + 8*halfSel
    const int rowBase = mt * 16 + halfSel * 8;
#pragma unroll
    for (int t = 0; t < 4; ++t) {
        v8f c = (t == 0) ? acc0 : (t == 1) ? acc1 : (t == 2) ? acc2 : acc3;
        int col = (nt0 + t) * 16 + lm;
#pragma unroll
        for (int v = 0; v < 8; ++v) {
            int row = rowBase + v;
            if (row < M) C[(long)row * N + col] = (_Float16)c[v];
        }
    }
}

// ---------- attention kernels ----------
__global__ void k_node_dots(const _Float16* __restrict__ hs, const float* __restrict__ a_src,
                            const float* __restrict__ a_dst, float* __restrict__ ssrc,
                            float* __restrict__ sdst) {
    int id = blockIdx.x * blockDim.x + threadIdx.x;
    if (id >= N_NODES * HEADS) return;
    int n = id / HEADS, hd = id % HEADS;
    const _Float16* hp = hs + (long)n * HC + hd * CH;
    float s1 = 0.f, s2 = 0.f;
    for (int c = 0; c < CH; ++c) {
        float v = (float)hp[c];
        s1 += v * a_src[hd * CH + c];
        s2 += v * a_dst[hd * CH + c];
    }
    ssrc[id] = s1; sdst[id] = s2;
}

// w_eff[k,h] = sum_c We[k, h*CH+c] * a_edge[h,c]   (EDGE_DIM x HEADS)
__global__ void k_weff(const float* __restrict__ We, const float* __restrict__ a_edge,
                       float* __restrict__ weff) {
    int id = blockIdx.x * blockDim.x + threadIdx.x;
    if (id >= EDGE_DIM * HEADS) return;
    int k = id / HEADS, hd = id % HEADS;
    float s = 0.f;
    for (int c = 0; c < CH; ++c) s += We[(long)k * HC + hd * CH + c] * a_edge[hd * CH + c];
    weff[id] = s;
}

__global__ void k_edge_logit(const float* __restrict__ ea, const int* __restrict__ src,
                             const int* __restrict__ dst, const float* __restrict__ ssrc,
                             const float* __restrict__ sdst, const float* __restrict__ weff,
                             float* __restrict__ logit) {
    int id = blockIdx.x * blockDim.x + threadIdx.x;
    if (id >= N_EDGES * HEADS) return;
    int e = id / HEADS, hd = id % HEADS;
    float el = 0.f;
    for (int k = 0; k < EDGE_DIM; ++k) el += ea[(long)e * EDGE_DIM + k] * weff[k * HEADS + hd];
    float l = ssrc[src[e] * HEADS + hd] + sdst[dst[e] * HEADS + hd] + el;
    logit[id] = (l > 0.f) ? l : 0.2f * l;
}

__global__ void k_init_neginf(unsigned* __restrict__ p, int n) {
    int id = blockIdx.x * blockDim.x + threadIdx.x;
    if (id < n) p[id] = enc_f32(-INFINITY);
}

__global__ void k_seg_max(const float* __restrict__ logit, const int* __restrict__ dst,
                          unsigned* __restrict__ mkey) {
    int id = blockIdx.x * blockDim.x + threadIdx.x;
    if (id >= N_EDGES * HEADS) return;
    int e = id / HEADS, hd = id % HEADS;
    atomicMax(&mkey[dst[e] * HEADS + hd], enc_f32(logit[id]));
}

__global__ void k_fin_max(const unsigned* __restrict__ mkey, float* __restrict__ mval, int n) {
    int id = blockIdx.x * blockDim.x + threadIdx.x;
    if (id >= n) return;
    float m = dec_f32(mkey[id]);
    mval[id] = isfinite(m) ? m : 0.0f;
}

__global__ void k_edge_p(const float* __restrict__ logit, const int* __restrict__ dst,
                         const float* __restrict__ mval, float* __restrict__ p,
                         float* __restrict__ denom) {
    int id = blockIdx.x * blockDim.x + threadIdx.x;
    if (id >= N_EDGES * HEADS) return;
    int e = id / HEADS, hd = id % HEADS;
    float pe = __expf(logit[id] - mval[dst[e] * HEADS + hd]);
    p[id] = pe;
    atomicAdd(&denom[dst[e] * HEADS + hd], pe);
}

// out[dst, c] += (alpha/HEADS) * hs[src, h, c]  — one block per edge, 8 heads x 32 lanes
__global__ void k_messages(const _Float16* __restrict__ hs, const int* __restrict__ src,
                           const int* __restrict__ dst, const float* __restrict__ p,
                           const float* __restrict__ denom, float* __restrict__ out) {
    int e = blockIdx.x;
    int hd = threadIdx.x >> 5, lane = threadIdx.x & 31;
    int s = src[e], d = dst[e];
    float alpha = p[e * HEADS + hd] / fmaxf(denom[d * HEADS + hd], 1e-16f);
    float scale = alpha * (1.0f / HEADS);
    const _Float16* hp = hs + (long)s * HC + hd * CH;
    float* op = out + (long)d * CH;
    for (int c = lane; c < CH; c += 32) atomicAdd(&op[c], scale * (float)hp[c]);
}

__global__ void k_add_bias(float* __restrict__ h, const float* __restrict__ b) {
    long id = (long)blockIdx.x * blockDim.x + threadIdx.x;
    if (id < (long)N_NODES * CH) h[id] += b[id % CH];
}

// ---------- graph norm ----------
__global__ void k_count(const int* __restrict__ batch, float* __restrict__ cnt) {
    int n = blockIdx.x * blockDim.x + threadIdx.x;
    if (n < N_NODES) atomicAdd(&cnt[batch[n]], 1.0f);
}
__global__ void k_gn_sum(const float* __restrict__ h, const int* __restrict__ batch,
                         float* __restrict__ gsum) {
    int n = blockIdx.x, c = threadIdx.x;
    atomicAdd(&gsum[batch[n] * CH + c], h[(long)n * CH + c]);
}
__global__ void k_div_cnt(const float* __restrict__ gsum, const float* __restrict__ cnt,
                          float* __restrict__ mean) {
    int id = blockIdx.x * blockDim.x + threadIdx.x;
    if (id >= N_GRAPHS * CH) return;
    mean[id] = gsum[id] / fmaxf(cnt[id / CH], 1.0f);
}
__global__ void k_gn_var(const float* __restrict__ h, const int* __restrict__ batch,
                         const float* __restrict__ mean, const float* __restrict__ gm,
                         float* __restrict__ gvar) {
    int n = blockIdx.x, c = threadIdx.x;
    int g = batch[n];
    float ctr = h[(long)n * CH + c] - mean[g * CH + c] * gm[c];
    atomicAdd(&gvar[g * CH + c], ctr * ctr);
}
__global__ void k_gn_apply(float* __restrict__ h, const int* __restrict__ batch,
                           const float* __restrict__ mean, const float* __restrict__ var,
                           const float* __restrict__ gw, const float* __restrict__ gb,
                           const float* __restrict__ gm) {
    int n = blockIdx.x, c = threadIdx.x;
    int g = batch[n];
    float ctr = h[(long)n * CH + c] - mean[g * CH + c] * gm[c];
    float o = gw[c] * ctr * rsqrtf(var[g * CH + c] + EPSV) + gb[c];
    h[(long)n * CH + c] = fmaxf(o, 0.0f);
}

// ---------- pooling + head ----------
__global__ void k_pool(const float* __restrict__ h, const int* __restrict__ batch,
                       float* __restrict__ gsum, unsigned* __restrict__ gmaxk) {
    int n = blockIdx.x, c = threadIdx.x;
    int g = batch[n];
    float v = h[(long)n * CH + c];
    atomicAdd(&gsum[g * CH + c], v);
    atomicMax(&gmaxk[g * CH + c], enc_f32(v));
}

__global__ void k_head(const float* __restrict__ gsum, const unsigned* __restrict__ gmaxk,
                       const float* __restrict__ cnt, const float* __restrict__ conv_w,
                       const float* __restrict__ conv_b, const float* __restrict__ fc_w,
                       const float* __restrict__ fc_b, float* __restrict__ out) {
    __shared__ float feat[2 * CH];
    __shared__ float partial[256];
    int g = blockIdx.x, tid = threadIdx.x;
    float mx = dec_f32(gmaxk[g * CH + tid]);
    feat[tid]      = isfinite(mx) ? mx : 0.0f;                         // gmax
    feat[CH + tid] = gsum[g * CH + tid] / fmaxf(cnt[g], 1.0f);         // gmean
    __syncthreads();

    float acc = 0.f;
    for (int idx = tid; idx < CONV_OUT * POOL_LEN; idx += 256) {
        int oc = idx / POOL_LEN, pp = idx % POOL_LEN;
        float z0 = conv_b[oc], z1 = conv_b[oc];
        int p0 = 2 * pp;
#pragma unroll
        for (int k = 0; k < KSIZE; ++k) {
            float w = conv_w[oc * KSIZE + k];
            z0 += feat[p0 + k] * w;
            z1 += feat[p0 + 1 + k] * w;
        }
        z0 = fmaxf(z0, 0.f); z1 = fmaxf(z1, 0.f);
        acc += fmaxf(z0, z1) * fc_w[idx];
    }
    partial[tid] = acc;
    __syncthreads();
    for (int s = 128; s > 0; s >>= 1) {
        if (tid < s) partial[tid] += partial[tid + s];
        __syncthreads();
    }
    if (tid == 0) out[g] = partial[0] + fc_b[0];
}

// ==================== host launch ====================
static inline size_t aln(size_t x) { return (x + 255) & ~(size_t)255; }

extern "C" void kernel_launch(void* const* d_in, const int* in_sizes, int n_in,
                              void* d_out, int out_size, void* d_ws, size_t ws_size,
                              hipStream_t stream) {
    const float* x       = (const float*)d_in[0];
    const float* ea      = (const float*)d_in[1];
    const int*   eidx    = (const int*)d_in[2];
    const int*   batch   = (const int*)d_in[4];
    const int*   src     = eidx;
    const int*   dst     = eidx + N_EDGES;
    const float* conv_w  = (const float*)d_in[32];
    const float* conv_b  = (const float*)d_in[33];
    const float* fc_w    = (const float*)d_in[34];
    const float* fc_b    = (const float*)d_in[35];
    float* out = (float*)d_out;

    // ---- workspace carve-up ----
    char* ws = (char*)d_ws;
    size_t off = 0;
    auto take = [&](size_t bytes) { char* p = ws + off; off += aln(bytes); return (void*)p; };
    _Float16* Apack = (_Float16*)take((size_t)MP * CH * 2);             // packed A (max Kp=256)
    _Float16* Bpack = (_Float16*)take((size_t)CH * HC * 2);             // packed W
    _Float16* hs    = (_Float16*)take((size_t)N_NODES * HC * 2);        // per-head features
    float* hbuf0    = (float*)take((size_t)N_NODES * CH * 4);
    float* hbuf1    = (float*)take((size_t)N_NODES * CH * 4);
    float* ssrc     = (float*)take((size_t)N_NODES * HEADS * 4);
    float* sdst     = (float*)take((size_t)N_NODES * HEADS * 4);
    float* logit    = (float*)take((size_t)N_EDGES * HEADS * 4);
    float* pbuf     = (float*)take((size_t)N_EDGES * HEADS * 4);
    unsigned* mkey  = (unsigned*)take((size_t)N_NODES * HEADS * 4);
    float* mval     = (float*)take((size_t)N_NODES * HEADS * 4);
    float* denom    = (float*)take((size_t)N_NODES * HEADS * 4);
    float* weff     = (float*)take((size_t)EDGE_DIM * HEADS * 4);
    float* cnt      = (float*)take((size_t)N_GRAPHS * 4);
    float* gsum     = (float*)take((size_t)N_GRAPHS * CH * 4);
    float* gvar     = (float*)take((size_t)N_GRAPHS * CH * 4);
    float* mean     = (float*)take((size_t)N_GRAPHS * CH * 4);
    float* var      = (float*)take((size_t)N_GRAPHS * CH * 4);
    unsigned* gmaxk = (unsigned*)take((size_t)N_GRAPHS * CH * 4);
    (void)ws_size; (void)n_in; (void)in_sizes; (void)out_size;

    const int NH = N_NODES * HEADS;
    const int EH = N_EDGES * HEADS;
    const int TB = 256;
    auto blk = [](long n, int t) { return (unsigned)((n + t - 1) / t); };

    // graph node counts (fixed across layers)
    hipMemsetAsync(cnt, 0, N_GRAPHS * 4, stream);
    k_count<<<blk(N_NODES, TB), TB, 0, stream>>>(batch, cnt);

    float* hin = hbuf0;
    float* hnew = hbuf1;

    for (int l = 0; l < 3; ++l) {
        const float* W      = (const float*)d_in[5 + 9 * l + 0];
        const float* We     = (const float*)d_in[5 + 9 * l + 1];
        const float* a_src  = (const float*)d_in[5 + 9 * l + 2];
        const float* a_dst  = (const float*)d_in[5 + 9 * l + 3];
        const float* a_edge = (const float*)d_in[5 + 9 * l + 4];
        const float* bias   = (const float*)d_in[5 + 9 * l + 5];
        const float* gw     = (const float*)d_in[5 + 9 * l + 6];
        const float* gb     = (const float*)d_in[5 + 9 * l + 7];
        const float* gm     = (const float*)d_in[5 + 9 * l + 8];

        int K  = (l == 0) ? IN_CH : CH;
        int KT = (l == 0) ? 1     : (CH / 32);   // k-tiles of 32
        const float* hsrc = (l == 0) ? x : hin;

        // pack operands into WMMA fragment order (f16), then WMMA GEMM -> hs (f16)
        long aThreads = (long)(MP >> 4) * KT * 32;
        long bThreads = (long)KT * (HC >> 4) * 32;
        k_pack_a<<<blk(aThreads, TB), TB, 0, stream>>>(hsrc, Apack, N_NODES, MP, K, KT);
        k_pack_b<<<blk(bThreads, TB), TB, 0, stream>>>(W, Bpack, K, KT, HC);
        k_wmma_gemm<<<dim3(HC / 64, MP / 64), 128, 0, stream>>>(Apack, Bpack, hs,
                                                                N_NODES, KT, HC);

        // attention scores
        k_node_dots<<<blk(NH, TB), TB, 0, stream>>>(hs, a_src, a_dst, ssrc, sdst);
        k_weff<<<1, EDGE_DIM * HEADS, 0, stream>>>(We, a_edge, weff);
        k_edge_logit<<<blk(EH, TB), TB, 0, stream>>>(ea, src, dst, ssrc, sdst, weff, logit);

        // segment softmax over dst
        k_init_neginf<<<blk(NH, TB), TB, 0, stream>>>(mkey, NH);
        k_seg_max<<<blk(EH, TB), TB, 0, stream>>>(logit, dst, mkey);
        k_fin_max<<<blk(NH, TB), TB, 0, stream>>>(mkey, mval, NH);
        hipMemsetAsync(denom, 0, (size_t)NH * 4, stream);
        k_edge_p<<<blk(EH, TB), TB, 0, stream>>>(logit, dst, mval, pbuf, denom);

        // message aggregation (mean over heads folded in), + bias
        hipMemsetAsync(hnew, 0, (size_t)N_NODES * CH * 4, stream);
        k_messages<<<N_EDGES, HEADS * 32, 0, stream>>>(hs, src, dst, pbuf, denom, hnew);
        k_add_bias<<<blk((long)N_NODES * CH, TB), TB, 0, stream>>>(hnew, bias);

        // graph norm + relu (in place on hnew)
        hipMemsetAsync(gsum, 0, (size_t)N_GRAPHS * CH * 4, stream);
        hipMemsetAsync(gvar, 0, (size_t)N_GRAPHS * CH * 4, stream);
        k_gn_sum<<<N_NODES, CH, 0, stream>>>(hnew, batch, gsum);
        k_div_cnt<<<blk(N_GRAPHS * CH, TB), TB, 0, stream>>>(gsum, cnt, mean);
        k_gn_var<<<N_NODES, CH, 0, stream>>>(hnew, batch, mean, gm, gvar);
        k_div_cnt<<<blk(N_GRAPHS * CH, TB), TB, 0, stream>>>(gvar, cnt, var);
        k_gn_apply<<<N_NODES, CH, 0, stream>>>(hnew, batch, mean, var, gw, gb, gm);

        // swap
        float* t = hin; hin = hnew; hnew = t;
    }

    // global pooling (gmax / gmean), conv1d + relu + maxpool + fc
    hipMemsetAsync(gsum, 0, (size_t)N_GRAPHS * CH * 4, stream);
    k_init_neginf<<<blk(N_GRAPHS * CH, TB), TB, 0, stream>>>(gmaxk, N_GRAPHS * CH);
    k_pool<<<N_NODES, CH, 0, stream>>>(hin, batch, gsum, gmaxk);
    k_head<<<N_GRAPHS, 256, 0, stream>>>(gsum, gmaxk, cnt, conv_w, conv_b, fc_w, fc_b, out);
}
